// KLRetrieval_46127948759328
// MI455X (gfx1250) — matrix operand order repaired and checked by reading, not compiled, not run on hardware
//
#include <hip/hip_runtime.h>

// ---------------------------------------------------------------------------
// Types for CDNA5 WMMA (gfx1250, wave32): D = A(16x32 bf16) * B(32x16 bf16) + C(f32)
// ---------------------------------------------------------------------------
typedef __bf16 bf16_t;
typedef bf16_t bf16x4 __attribute__((ext_vector_type(4)));
typedef bf16_t bf16x8 __attribute__((ext_vector_type(8)));
typedef bf16_t v16bf  __attribute__((ext_vector_type(16)));
typedef float  v8f    __attribute__((ext_vector_type(8)));

union Frag { v16bf v; bf16x8 h2[2]; };

// ---------------------------------------------------------------------------
// Generic strided/batched/gathered bf16-WMMA GEMM:
//   C[b,h][m,n] = scale * sum_k A(m,k)*B(k,n) + bias[n] + Res(m,n)
//   A(m,k) = A + b*sAb + h*sAh + rowOf(m)*ldA + k          (rowOf via rowIdx gather)
//   B(k,n) = B + b*sBb + h*sBh + k*ldBk + n*ldBn
//   B_TRANS=true assumes ldBk==1 (K contiguous), else ldBn==1 (N contiguous).
// All M,N multiples of 128 and K multiples of 32 in this workload (no edge guards).
// ---------------------------------------------------------------------------
struct GemmP {
  const float* A; long ldA, sAb, sAh; const int* rowIdx;
  const float* B; long ldBk, ldBn, sBb, sBh;
  float* C; long ldC, sCb, sCh;
  const float* bias;
  const float* Res; long ldR, sRb, sRh; const int* resIdx;
  int M, N, Kd, hbits;
  float scale;
};

#define BM 128
#define BN 128
#define BK 32
#define LDA_S 40   // 32 + 8 pad bf16 -> 80B rows (16B aligned), kills bank conflicts
#define LDB_S 136  // 128 + 8 pad bf16 -> 272B rows (16B aligned)

template<bool GATHER_A, bool B_TRANS>
__global__ void __launch_bounds__(256, 2) wmma_gemm_bf16(GemmP p)
{
  // ping-pong LDS: ~38 KB total (WGP has 320 KB)
  __shared__ alignas(16) bf16_t As[2][BM * LDA_S];
  __shared__ alignas(16) bf16_t Bs[2][BK * LDB_S];

  const int tid  = threadIdx.x;
  const int lane = tid & 31;
  const int wv   = tid >> 5;
  const int wM   = wv & 3;       // 4 waves along M, 32 rows each
  const int wN   = wv >> 2;      // 2 waves along N, 64 cols each
  const int r    = lane & 15;
  const int hl   = lane >> 4;    // half-wave select

  const long bz = blockIdx.z;
  const long bb = bz >> p.hbits;
  const long hh = bz & ((1L << p.hbits) - 1L);

  const float* Ab = p.A + bb * p.sAb + hh * p.sAh;
  const float* Bb = p.B + bb * p.sBb + hh * p.sBh;
  float*       Cb = p.C + bb * p.sCb + hh * p.sCh;

  const int m0 = blockIdx.y * BM;
  const int n0 = blockIdx.x * BN;

  // ---- K-invariant staging coordinates (gather rows hoisted out of K loop) ----
  long aGlob[4]; int aLds[4];
  #pragma unroll
  for (int i = 0; i < 4; ++i) {
    int q  = tid + i * 256;            // 1024 float4 groups over 128x32
    int am = q >> 3, ak = (q & 7) * 4;
    long row = GATHER_A ? (long)p.rowIdx[m0 + am] : (long)(m0 + am);
    aGlob[i] = row * p.ldA + ak;
    aLds[i]  = am * LDA_S + ak;
  }
  long bGlob[4]; int bLds[4];
  #pragma unroll
  for (int i = 0; i < 4; ++i) {
    int q = tid + i * 256;
    if (B_TRANS) {                     // vectorize along K (contiguous, ldBk==1)
      int kg = (q & 7) * 4, bn = q >> 3;
      bGlob[i] = (long)kg + (long)(n0 + bn) * p.ldBn;
      bLds[i]  = kg * LDB_S + bn;
    } else {                           // vectorize along N (contiguous, ldBn==1)
      int bn = (q & 31) * 4, bk = q >> 5;
      bGlob[i] = (long)bk * p.ldBk + (long)(n0 + bn);
      bLds[i]  = bk * LDB_S + bn;
    }
  }

  float4 aReg[4], bReg[4];

  auto prefetch = [&](int k0) {
    #pragma unroll
    for (int i = 0; i < 4; ++i)
      aReg[i] = *(const float4*)(Ab + aGlob[i] + k0);        // global_load_b128
    const long bStep = B_TRANS ? (long)k0 : (long)k0 * p.ldBk;
    #pragma unroll
    for (int i = 0; i < 4; ++i)
      bReg[i] = *(const float4*)(Bb + bGlob[i] + bStep);
  };

  auto commit = [&](int buf) {
    #pragma unroll
    for (int i = 0; i < 4; ++i) {
      bf16x4 v = { (bf16_t)aReg[i].x, (bf16_t)aReg[i].y,
                   (bf16_t)aReg[i].z, (bf16_t)aReg[i].w };
      *(bf16x4*)&As[buf][aLds[i]] = v;                        // ds_store_b64
    }
    #pragma unroll
    for (int i = 0; i < 4; ++i) {
      if (B_TRANS) {                   // 4 consecutive K rows of one column
        Bs[buf][bLds[i]            ] = (bf16_t)bReg[i].x;
        Bs[buf][bLds[i] +     LDB_S] = (bf16_t)bReg[i].y;
        Bs[buf][bLds[i] + 2 * LDB_S] = (bf16_t)bReg[i].z;
        Bs[buf][bLds[i] + 3 * LDB_S] = (bf16_t)bReg[i].w;
      } else {
        bf16x4 v = { (bf16_t)bReg[i].x, (bf16_t)bReg[i].y,
                     (bf16_t)bReg[i].z, (bf16_t)bReg[i].w };
        *(bf16x4*)&Bs[buf][bLds[i]] = v;
      }
    }
  };

  v8f acc[2][4] = {};

  auto compute = [&](int buf) {
    // Fragments per ISA 7.12.2 striping; ds_load_b128 pairs
    Frag fa[2], fb[4];
    #pragma unroll
    for (int mf = 0; mf < 2; ++mf) {
      const bf16_t* ap = &As[buf][(wM * 32 + mf * 16 + r) * LDA_S];
      fa[mf].h2[0] = *(const bf16x8*)(ap + 8 * hl);           // K = 8h..8h+7
      fa[mf].h2[1] = *(const bf16x8*)(ap + 16 + 8 * hl);      // K = 16+8h..
    }
    const bf16_t* bp = &Bs[buf][wN * 64 + r]; // B column n=r, K = 8h..8h+7, 16+8h..
    #pragma unroll
    for (int nf = 0; nf < 4; ++nf) {
      #pragma unroll
      for (int s = 0; s < 8; ++s) {
        fb[nf].v[s]     = bp[(8 * hl + s) * LDB_S + nf * 16];
        fb[nf].v[s + 8] = bp[(16 + 8 * hl + s) * LDB_S + nf * 16];
      }
    }
    #pragma unroll
    for (int mf = 0; mf < 2; ++mf)
      #pragma unroll
      for (int nf = 0; nf < 4; ++nf)
        acc[mf][nf] = __builtin_amdgcn_wmma_f32_16x16x32_bf16(
            false, fa[mf].v, false, fb[nf].v, (short)0, acc[mf][nf], false, false);
  };

  // ---- software-pipelined main loop: 1 barrier / K-step ----
  const int nsteps = p.Kd / BK;
  prefetch(0);
  commit(0);
  __syncthreads();
  for (int s = 0; s < nsteps; ++s) {
    const int buf = s & 1;
    if (s + 1 < nsteps) prefetch((s + 1) * BK);   // global loads overlap WMMAs
    compute(buf);
    if (s + 1 < nsteps) commit(buf ^ 1);          // write other buffer: no race
    __syncthreads();
  }

  // ---- Epilogue: scale, bias, (gathered) residual ----
  const float* Rb = p.Res ? (p.Res + bb * p.sRb + hh * p.sRh) : nullptr;
  #pragma unroll
  for (int mf = 0; mf < 2; ++mf) {
    #pragma unroll
    for (int nf = 0; nf < 4; ++nf) {
      const int rowB = m0 + wM * 32 + mf * 16 + 8 * hl;
      const int col  = n0 + wN * 64 + nf * 16 + r;
      #pragma unroll
      for (int j = 0; j < 8; ++j) {
        const int row = rowB + j;
        float v = acc[mf][nf][j] * p.scale;
        if (p.bias) v += p.bias[col];
        if (Rb) {
          long rr = p.resIdx ? (long)p.resIdx[row] : (long)row;
          v += Rb[rr * p.ldR + col];
        }
        Cb[(long)row * p.ldC + col] = v;
      }
    }
  }
}

// ---------------------------------------------------------------------------
// Small kernels: MLP dense layers, loss/argmax, gather-index build, row softmax
// ---------------------------------------------------------------------------
__global__ void dense_act_kernel(const float* __restrict__ X,
                                 const float* __restrict__ W,
                                 const float* __restrict__ b,
                                 float* __restrict__ Y,
                                 int M, int Kd, int Nn, int act)
{
  int i = blockIdx.x * blockDim.x + threadIdx.x;
  if (i >= M * Nn) return;
  int m = i / Nn, n = i % Nn;
  const float* xr = X + (long)m * Kd;
  float acc = b[n];
  for (int k = 0; k < Kd; ++k) acc = fmaf(xr[k], W[(long)k * Nn + n], acc);
  if (act == 1) acc = fmaxf(acc, 0.0f);
  else if (act == 2) acc = 1.0f / (1.0f + expf(-acc));
  Y[i] = acc;
}

__global__ void loss_argmax_kernel(const float* __restrict__ logits,
                                   const int* __restrict__ label,
                                   float* __restrict__ lossOut,
                                   int* __restrict__ cls)
{
  __shared__ float partial[16];
  int b = threadIdx.x;
  if (b < 16) {
    const float* l = logits + b * 12;
    float mx = l[0]; int am = 0;
    for (int j = 1; j < 12; ++j) if (l[j] > mx) { mx = l[j]; am = j; }
    cls[b] = am;
    float s = 0.0f;
    for (int j = 0; j < 12; ++j) s += expf(l[j] - mx);
    float logp = l[label[b]] - mx - logf(s);
    partial[b] = -logp;
  }
  __syncthreads();
  if (threadIdx.x == 0) {
    float t = 0.0f;
    for (int j = 0; j < 16; ++j) t += partial[j];
    *lossOut = t / 16.0f;
  }
}

__global__ void build_gather_kernel(const int* __restrict__ cls,
                                    const int* __restrict__ le,
                                    const int* __restrict__ re,
                                    const int* __restrict__ ra,
                                    int* __restrict__ gidx,
                                    int* __restrict__ gidxRel)
{
  int m = blockIdx.x * 256 + threadIdx.x;
  if (m < 8192) {                    // embed rows: [B=16][2K=512]
    int b = m >> 9, j = m & 511;
    int c = cls[b];
    gidx[m] = (j < 256) ? le[c * 256 + j] : re[c * 256 + (j - 256)];
  }
  if (m < 4096) {                    // rel rows: [B=16][K=256]
    int b = m >> 8, j = m & 255;
    gidxRel[m] = ra[cls[b] * 256 + j];
  }
}

__global__ void softmax_rows_kernel(float* __restrict__ X, int L)
{
  __shared__ float red[256];
  float* x = X + (long)blockIdx.x * L;
  int t = threadIdx.x;
  float mx = -1e30f;
  for (int i = t; i < L; i += 256) mx = fmaxf(mx, x[i]);
  red[t] = mx; __syncthreads();
  for (int s = 128; s > 0; s >>= 1) { if (t < s) red[t] = fmaxf(red[t], red[t + s]); __syncthreads(); }
  mx = red[0]; __syncthreads();
  float sum = 0.0f;
  for (int i = t; i < L; i += 256) { float e = expf(x[i] - mx); x[i] = e; sum += e; }
  red[t] = sum; __syncthreads();
  for (int s = 128; s > 0; s >>= 1) { if (t < s) red[t] += red[t + s]; __syncthreads(); }
  float inv = 1.0f / red[0];
  for (int i = t; i < L; i += 256) x[i] *= inv;
}

// ---------------------------------------------------------------------------
// Host-side orchestration
// ---------------------------------------------------------------------------
static inline GemmP gp(const float* A, long ldA, const float* Bm, long ldBk, long ldBn,
                       float* C, long ldC, int M, int N, int Kd)
{
  GemmP p{};
  p.A = A; p.ldA = ldA; p.B = Bm; p.ldBk = ldBk; p.ldBn = ldBn;
  p.C = C; p.ldC = ldC; p.M = M; p.N = N; p.Kd = Kd;
  p.hbits = 0; p.scale = 1.0f;
  return p;
}

static inline void launch_gemm(bool gather, bool btrans, const GemmP& p, int Z,
                               hipStream_t s)
{
  dim3 g(p.N / BN, p.M / BM, Z), b(256);
  if (gather) {
    if (btrans) wmma_gemm_bf16<true,  true ><<<g, b, 0, s>>>(p);
    else        wmma_gemm_bf16<true,  false><<<g, b, 0, s>>>(p);
  } else {
    if (btrans) wmma_gemm_bf16<false, true ><<<g, b, 0, s>>>(p);
    else        wmma_gemm_bf16<false, false><<<g, b, 0, s>>>(p);
  }
}

extern "C" void kernel_launch(void* const* d_in, const int* in_sizes, int n_in,
                              void* d_out, int out_size, void* d_ws, size_t ws_size,
                              hipStream_t stream)
{
  (void)in_sizes; (void)n_in; (void)out_size; (void)ws_size;

  const float* x    = (const float*)d_in[0];
  const float* img  = (const float*)d_in[1];
  const int*   lbl  = (const int*)  d_in[2];
  const float* ent  = (const float*)d_in[3];
  const float* rtab = (const float*)d_in[4];
  const int*   le   = (const int*)  d_in[5];
  const int*   re   = (const int*)  d_in[6];
  const int*   ra   = (const int*)  d_in[7];
  const float* w1 = (const float*)d_in[8];  const float* b1 = (const float*)d_in[9];
  const float* w2 = (const float*)d_in[10]; const float* b2 = (const float*)d_in[11];
  const float* w3 = (const float*)d_in[12]; const float* b3 = (const float*)d_in[13];
  const float* Wq = (const float*)d_in[14]; const float* bq = (const float*)d_in[15];
  const float* Wk = (const float*)d_in[16]; const float* bk = (const float*)d_in[17];
  const float* Wv = (const float*)d_in[18]; const float* bv = (const float*)d_in[19];
  const float* Wo = (const float*)d_in[20]; const float* bo = (const float*)d_in[21];
  const float* Wf = (const float*)d_in[22]; const float* bff= (const float*)d_in[23];
  float* out = (float*)d_out;

  // Workspace carve-out
  char* base = (char*)d_ws;
  size_t off = 0;
  auto take = [&](size_t bytes) -> char* {
    char* pr = base + off;
    off = (off + bytes + 255) & ~(size_t)255;
    return pr;
  };
  float* h1     = (float*)take(16L * 1024 * 4);
  float* hh2    = (float*)take(16L * 256 * 4);
  float* logits = (float*)take(16L * 12 * 4);
  int*   cls    = (int*)  take(16L * 4);
  int*   gidx   = (int*)  take(8192L * 4);
  int*   gidxR  = (int*)  take(4096L * 4);
  float* Q   = (float*)take(16L * 512 * 2048 * 4);    // 64 MB (reused as O)
  float* Kp  = (float*)take(16L * 256 * 2048 * 4);    // 32 MB
  float* Vp  = (float*)take(16L * 256 * 2048 * 4);    // 32 MB
  float* S   = (float*)take(16L * 8 * 512 * 256 * 4); // 64 MB scores/attn
  float* KF  = (float*)take(16L * 512 * 2048 * 4);    // 64 MB
  float* S2  = (float*)take(16L * 256 * 512 * 4);     // 8 MB
  float* CT  = (float*)take(16L * 256 * 2048 * 4);    // 32 MB
  float* O = Q; // scores consume Q before O is written; safe stream-ordered reuse

  // ---- MLP classifier + loss + argmax ----
  dense_act_kernel<<<64, 256, 0, stream>>>(x,   w1, b1, h1,     16, 4096, 1024, 1);
  dense_act_kernel<<<16, 256, 0, stream>>>(h1,  w2, b2, hh2,    16, 1024, 256,  1);
  dense_act_kernel<<<1,  256, 0, stream>>>(hh2, w3, b3, logits, 16, 256,  12,   2);
  loss_argmax_kernel<<<1, 32, 0, stream>>>(logits, lbl, out + 8388608L, cls);
  build_gather_kernel<<<32, 256, 0, stream>>>(cls, le, re, ra, gidx, gidxR);

  // ---- Q = gather(entity, embed rows) @ Wq + bq      [8192 x 2048] ----
  { GemmP p = gp(ent, 2048, Wq, 2048, 1, Q, 2048, 8192, 2048, 2048);
    p.rowIdx = gidx; p.bias = bq;
    launch_gemm(true, false, p, 1, stream); }
  // ---- K = gather(rela) @ Wk + bk; V likewise       [4096 x 2048] ----
  { GemmP p = gp(rtab, 2048, Wk, 2048, 1, Kp, 2048, 4096, 2048, 2048);
    p.rowIdx = gidxR; p.bias = bk;
    launch_gemm(true, false, p, 1, stream); }
  { GemmP p = gp(rtab, 2048, Wv, 2048, 1, Vp, 2048, 4096, 2048, 2048);
    p.rowIdx = gidxR; p.bias = bv;
    launch_gemm(true, false, p, 1, stream); }

  // ---- scores[b,h] = Q_h @ K_h^T / 16   (Z = 128 (b,h) pairs) ----
  { GemmP p = gp(Q, 2048, Kp, /*ldBk=*/1, /*ldBn=*/2048, S, 256, 512, 256, 256);
    p.sAb = 512L * 2048; p.sAh = 256;
    p.sBb = 256L * 2048; p.sBh = 256;
    p.sCb = 8L * 512 * 256; p.sCh = 512L * 256;
    p.hbits = 3; p.scale = 0.0625f;
    launch_gemm(false, true, p, 128, stream); }
  softmax_rows_kernel<<<65536, 256, 0, stream>>>(S, 256);

  // ---- O[b,:,h*256+ :] = attn @ V_h ----
  { GemmP p = gp(S, 256, Vp, 2048, 1, O, 2048, 512, 256, 256);
    p.sAb = 8L * 512 * 256; p.sAh = 512L * 256;
    p.sBb = 256L * 2048;    p.sBh = 256;
    p.sCb = 512L * 2048;    p.sCh = 256;
    p.hbits = 3;
    launch_gemm(false, false, p, 128, stream); }

  // ---- kf = O @ Wo + bo + embed (gathered residual) [8192 x 2048] ----
  { GemmP p = gp(O, 2048, Wo, 2048, 1, KF, 2048, 8192, 2048, 2048);
    p.bias = bo; p.Res = ent; p.ldR = 2048; p.resIdx = gidx;
    launch_gemm(false, false, p, 1, stream); }

  // ---- s2[b] = img[b] @ kf[b]^T / sqrt(2048)   (Z = 16) ----
  { GemmP p = gp(img, 2048, KF, /*ldBk=*/1, /*ldBn=*/2048, S2, 512, 256, 512, 2048);
    p.sAb = 256L * 2048; p.sBb = 512L * 2048; p.sCb = 256L * 512;
    p.scale = 0.022097086912079608f;
    launch_gemm(false, true, p, 16, stream); }
  softmax_rows_kernel<<<4096, 256, 0, stream>>>(S2, 512);

  // ---- ctx[b] = a2[b] @ kf[b] ----
  { GemmP p = gp(S2, 512, KF, 2048, 1, CT, 2048, 256, 2048, 512);
    p.sAb = 256L * 512; p.sBb = 512L * 2048; p.sCb = 256L * 2048;
    launch_gemm(false, false, p, 16, stream); }

  // ---- out = ctx @ Wf + bf + imageFeature  ->  d_out [4096 x 2048] ----
  { GemmP p = gp(CT, 2048, Wf, 2048, 1, out, 2048, 4096, 2048, 2048);
    p.bias = bff; p.Res = img; p.ldR = 2048;
    launch_gemm(false, false, p, 1, stream); }
}
